// ScreenplayGPT_6768868458993
// MI455X (gfx1250) — matrix operand order repaired and checked by reading, not compiled, not run on hardware
//
#include <hip/hip_runtime.h>
#include <math.h>
#include <stdint.h>

// ---------------------------------------------------------------------------
// Types for WMMA fragments (wave32, v_wmma_f32_16x16x32_bf16)
// ---------------------------------------------------------------------------
typedef __attribute__((ext_vector_type(16))) __bf16          v16bf;
typedef __attribute__((ext_vector_type(8)))  float           v8f;
typedef __attribute__((ext_vector_type(8)))  unsigned short  us8_t;

union Frag { us8_t h[2]; v16bf v; };

#define DEV __device__ __forceinline__

// round-to-nearest-even fp32 -> bf16 (bit pattern only; bf16 only feeds WMMA)
DEV unsigned short f2bf(float f) {
  unsigned int u = __float_as_uint(f);
  u += 0x7FFFu + ((u >> 16) & 1u);
  return (unsigned short)(u >> 16);
}

// ---------------------------------------------------------------------------
// Embedding: x[t,d] = wte[idx[t], d] + wpe[t, d]
// ---------------------------------------------------------------------------
__global__ void embed_k(const int* __restrict__ idx,
                        const float* __restrict__ wte,
                        const float* __restrict__ wpe,
                        float* __restrict__ x, int S, int Dm) {
  size_t i = (size_t)blockIdx.x * blockDim.x + threadIdx.x;
  if (i >= (size_t)S * Dm) return;
  int t = (int)(i / Dm);
  int d = (int)(i % Dm);
  int tok = idx[t];
  x[i] = wte[(size_t)tok * Dm + d] + wpe[(size_t)t * Dm + d];
}

// ---------------------------------------------------------------------------
// Elementwise fp32 -> bf16 convert (wte for weight-tied lm_head B operand)
// ---------------------------------------------------------------------------
__global__ void conv_k(const float* __restrict__ in,
                       unsigned short* __restrict__ out, size_t n) {
  size_t i = (size_t)blockIdx.x * blockDim.x + threadIdx.x;
  if (i < n) out[i] = f2bf(in[i]);
}

// ---------------------------------------------------------------------------
// Transpose-convert: in [K,N] fp32 row-major -> out [N,K] bf16 row-major
// (all GEMMs run "NT" so every B-fragment load is contiguous)
// ---------------------------------------------------------------------------
__global__ void tconv_k(const float* __restrict__ in,
                        unsigned short* __restrict__ out, int K, int N) {
  size_t i = (size_t)blockIdx.x * blockDim.x + threadIdx.x;
  if (i >= (size_t)K * N) return;
  int n = (int)(i / K);
  int k = (int)(i % K);
  out[i] = f2bf(in[(size_t)k * N + n]);
}

// ---------------------------------------------------------------------------
// LayerNorm (fp32 in) -> bf16 out, one 256-thread block per row
// ---------------------------------------------------------------------------
__global__ void __launch_bounds__(256)
layernorm_k(const float* __restrict__ x, const float* __restrict__ w,
            const float* __restrict__ b, unsigned short* __restrict__ out,
            int Dm) {
  __shared__ float red[256];
  const int tid = threadIdx.x;
  const int row = blockIdx.x;
  const float* xr = x + (size_t)row * Dm;

  float s = 0.f;
  for (int i = tid; i < Dm; i += 256) s += xr[i];
  red[tid] = s; __syncthreads();
  for (int off = 128; off > 0; off >>= 1) {
    if (tid < off) red[tid] += red[tid + off];
    __syncthreads();
  }
  float mu = red[0] / (float)Dm;
  __syncthreads();

  float v = 0.f;
  for (int i = tid; i < Dm; i += 256) { float d = xr[i] - mu; v += d * d; }
  red[tid] = v; __syncthreads();
  for (int off = 128; off > 0; off >>= 1) {
    if (tid < off) red[tid] += red[tid + off];
    __syncthreads();
  }
  float inv = rsqrtf(red[0] / (float)Dm + 1e-5f);

  unsigned short* orow = out + (size_t)row * Dm;
  for (int i = tid; i < Dm; i += 256)
    orow[i] = f2bf((xr[i] - mu) * inv * w[i] + b[i]);
}

// ---------------------------------------------------------------------------
// Causal softmax: scores fp32 [S,S] (row i valid for j<=i) -> probs bf16 [S,S]
// ---------------------------------------------------------------------------
__global__ void __launch_bounds__(256)
softmax_causal_k(const float* __restrict__ sc,
                 unsigned short* __restrict__ pb, int S) {
  __shared__ float red[256];
  const int tid = threadIdx.x;
  const int i = blockIdx.x;
  const int len = i + 1;
  const float* r = sc + (size_t)i * S;

  float m = -3.4e38f;
  for (int j = tid; j < len; j += 256) m = fmaxf(m, r[j]);
  red[tid] = m; __syncthreads();
  for (int off = 128; off > 0; off >>= 1) {
    if (tid < off) red[tid] = fmaxf(red[tid], red[tid + off]);
    __syncthreads();
  }
  m = red[0];
  __syncthreads();

  float s = 0.f;
  for (int j = tid; j < len; j += 256) s += expf(r[j] - m);
  red[tid] = s; __syncthreads();
  for (int off = 128; off > 0; off >>= 1) {
    if (tid < off) red[tid] += red[tid + off];
    __syncthreads();
  }
  float inv = 1.0f / red[0];

  unsigned short* prow = pb + (size_t)i * S;
  for (int j = tid; j < S; j += 256) {
    float p = (j < len) ? expf(r[j] - m) * inv : 0.f;
    prow[j] = f2bf(p);
  }
}

// ---------------------------------------------------------------------------
// Split QKV fp32 [S,3D] -> q,k bf16 [H,S,HD]; v transposed to [H,HD,S]
// ---------------------------------------------------------------------------
__global__ void split_qkv_k(const float* __restrict__ qkv,
                            unsigned short* __restrict__ q,
                            unsigned short* __restrict__ k,
                            unsigned short* __restrict__ vT,
                            int S, int Dm, int HDm) {
  size_t i = (size_t)blockIdx.x * blockDim.x + threadIdx.x;
  if (i >= (size_t)S * Dm) return;
  int s = (int)(i / Dm);
  int d = (int)(i % Dm);
  int h = d / HDm, hd = d % HDm;
  const float* row = qkv + (size_t)s * 3 * Dm;
  q [((size_t)h * S + s) * HDm + hd] = f2bf(row[d]);
  k [((size_t)h * S + s) * HDm + hd] = f2bf(row[Dm + d]);
  vT[((size_t)h * HDm + hd) * S + s] = f2bf(row[2 * Dm + d]);
}

// ---------------------------------------------------------------------------
// Epilogue helper (alpha -> bias -> gelu -> residual -> fp32 / bf16)
// ---------------------------------------------------------------------------
DEV void epilogue_tile(const v8f& acc, int rbase, int col, int ldc, float alpha,
                       float bv, int gelu_flag,
                       const float* __restrict__ resid,
                       float* __restrict__ Cf, unsigned short* __restrict__ Cbf) {
#pragma unroll
  for (int r = 0; r < 8; ++r) {
    const size_t off = (size_t)(rbase + r) * ldc + col;
    float v = acc[r] * alpha + bv;
    if (gelu_flag) v = 0.5f * v * (1.0f + erff(v * 0.70710678118654752f));
    if (resid) v += resid[off];
    if (Cf)  Cf[off]  = v;
    if (Cbf) Cbf[off] = f2bf(v);
  }
}

// ---------------------------------------------------------------------------
// GEMM v1 (small-N path): one wave per 16x16 tile, direct global loads.
// ---------------------------------------------------------------------------
__global__ void __launch_bounds__(256)
gemm_nt_k(const unsigned short* __restrict__ A,
          const unsigned short* __restrict__ B,
          const float* __restrict__ bias,
          const float* __restrict__ resid,
          float* __restrict__ Cf,
          unsigned short* __restrict__ Cbf,
          int M, int N, int K, int lda, int ldb, int ldc,
          float alpha, int gelu_flag) {
  const int lane = threadIdx.x & 31;
  const int wave = threadIdx.x >> 5;
  const int n0   = (blockIdx.x * 8 + wave) * 16;
  const int m0   = blockIdx.y * 16;
  if (n0 >= N || m0 >= M) return;          // wave-uniform

  const int half = lane >> 4;
  const int lrow = lane & 15;

  const unsigned short* arow = A + (size_t)(m0 + lrow) * lda + (half << 4);
  const unsigned short* brow = B + (size_t)(n0 + lrow) * ldb + (half << 4);

  v8f acc = {};
  for (int k0 = 0; k0 < K; k0 += 32) {
    Frag a, b;
    const us8_t* pa = (const us8_t*)(arow + k0);
    a.h[0] = pa[0];
    a.h[1] = pa[1];
    const us8_t* pb = (const us8_t*)(brow + k0);
    b.h[0] = pb[0];
    b.h[1] = pb[1];
    __builtin_prefetch((const void*)(arow + k0 + 256), 0, 1);
    __builtin_prefetch((const void*)(brow + k0 + 256), 0, 1);
    acc = __builtin_amdgcn_wmma_f32_16x16x32_bf16(
        false, a.v, false, b.v, (short)0, acc, false, false);
  }

  const int col = n0 + lrow;
  float bv = bias ? bias[col] : 0.f;
  epilogue_tile(acc, m0 + (half << 3), col, ldc, alpha, bv, gelu_flag,
                resid, Cf, Cbf);
}

// ---------------------------------------------------------------------------
// GEMM v2 (main path): 256 threads = 8 waves as 2(M) x 4(N).
//   Block tile: 64 rows x 256 cols. Wave tile: 32 rows x 64 cols
//   = 2x4 WMMA tiles = 8 accumulators, A frags reused 4x, B frags 2x.
// A block-tile (64x32 bf16 = 4KB) is staged into LDS via the gfx1250 async
// copy path (global_load_async_to_lds_b128, ASYNCcnt) with DOUBLE BUFFERING:
// the stage for k+32 is issued before computing on k, and completion of the
// current buffer is guaranteed with s_wait_asynccnt 1 (async loads complete
// in order). B streams straight to registers (the scheduler pipelines those
// with LOADcnt skewing on its own).
// ---------------------------------------------------------------------------
__global__ void __launch_bounds__(256)
gemm_nt_v2_k(const unsigned short* __restrict__ A,
             const unsigned short* __restrict__ B,
             const float* __restrict__ bias,
             const float* __restrict__ resid,
             float* __restrict__ Cf,
             unsigned short* __restrict__ Cbf,
             int M, int N, int K, int lda, int ldb, int ldc,
             float alpha, int gelu_flag) {
  __shared__ __align__(16) unsigned short aTile[2][64 * 32];

  const int tid  = threadIdx.x;
  const int lane = tid & 31;
  const int wave = tid >> 5;
  const int wm   = wave >> 2;          // 0..1
  const int wn   = wave & 3;           // 0..3

  const int mblk = blockIdx.y * 64;
  const int nblk = blockIdx.x * 256;

  const int half = lane >> 4;
  const int lrow = lane & 15;

  // --- async A staging: thread t copies 16B: row tid>>2, quarter tid&3 ---
  const int arow = tid >> 2;           // 0..63
  const int aq   = tid & 3;            // 0..3
  const unsigned short* aG = A + (size_t)(mblk + arow) * lda + aq * 8;
  unsigned aL[2];
  aL[0] = (unsigned)(unsigned long long)(uintptr_t)(&aTile[0][arow * 32 + aq * 8]);
  aL[1] = (unsigned)(unsigned long long)(uintptr_t)(&aTile[1][arow * 32 + aq * 8]);

  // --- B pointers for 4 n-subtiles (col clamped so OOB tiles read safely;
  //     stores are guarded by the real column) ---
  const unsigned short* bptr[4];
  for (int nt = 0; nt < 4; ++nt) {
    int c0 = nblk + wn * 64 + nt * 16;
    if (c0 > N - 16) c0 = N - 16;
    bptr[nt] = B + (size_t)(c0 + lrow) * ldb + (half << 4);
  }

  // --- A fragment read offsets within a buffer (row pitch 32 ushorts) ---
  int aOff[2];
  for (int mt = 0; mt < 2; ++mt)
    aOff[mt] = (wm * 32 + mt * 16 + lrow) * 32 + (half << 4);

  v8f acc[2][4] = {};

  // prologue: stage k=0 into buffer 0
  {
    unsigned long long ga = (unsigned long long)(uintptr_t)aG;
    asm volatile("global_load_async_to_lds_b128 %0, %1, off"
                 :: "v"(aL[0]), "v"(ga) : "memory");
  }

  for (int k0 = 0; k0 < K; k0 += 32) {
    const int cur = (k0 >> 5) & 1;
    if (k0 + 32 < K) {
      // issue next stage into the other buffer, then wait only for current
      unsigned long long ga = (unsigned long long)(uintptr_t)(aG + k0 + 32);
      asm volatile("global_load_async_to_lds_b128 %0, %1, off"
                   :: "v"(aL[cur ^ 1]), "v"(ga) : "memory");
      asm volatile("s_wait_asynccnt 1" ::: "memory");
    } else {
      asm volatile("s_wait_asynccnt 0" ::: "memory");
    }
    __syncthreads();

    Frag a[2], b[4];
#pragma unroll
    for (int mt = 0; mt < 2; ++mt) {
      const us8_t* pa = (const us8_t*)(&aTile[cur][aOff[mt]]);
      a[mt].h[0] = pa[0];          // K = k0 + half*16 + [0..7]
      a[mt].h[1] = pa[1];          // K = k0 + half*16 + [8..15]
    }
#pragma unroll
    for (int nt = 0; nt < 4; ++nt) {
      const us8_t* pb = (const us8_t*)(bptr[nt] + k0);
      b[nt].h[0] = pb[0];          // K = k0 + half*16 + [0..7]
      b[nt].h[1] = pb[1];          // K = k0 + half*16 + [8..15]
      __builtin_prefetch((const void*)(bptr[nt] + k0 + 256), 0, 1);
    }
#pragma unroll
    for (int mt = 0; mt < 2; ++mt)
#pragma unroll
      for (int nt = 0; nt < 4; ++nt)
        acc[mt][nt] = __builtin_amdgcn_wmma_f32_16x16x32_bf16(
            false, a[mt].v, false, b[nt].v, (short)0, acc[mt][nt], false, false);
    __syncthreads();   // readers done before next DMA overwrites this buffer
  }

  // --- epilogue (store guards use the true, unclamped column) ---
  for (int mt = 0; mt < 2; ++mt) {
    const int rbase = mblk + wm * 32 + mt * 16 + (half << 3);
    for (int nt = 0; nt < 4; ++nt) {
      const int c0 = nblk + wn * 64 + nt * 16;
      if (c0 >= N) continue;                   // wave-uniform
      const int col = c0 + lrow;
      float bv = bias ? bias[col] : 0.f;
      epilogue_tile(acc[mt][nt], rbase, col, ldc, alpha, bv, gelu_flag,
                    resid, Cf, Cbf);
    }
  }
}

// ---------------------------------------------------------------------------
// Host side
// ---------------------------------------------------------------------------
static inline void launch_gemm(hipStream_t s,
                               const unsigned short* A, const unsigned short* B,
                               const float* bias, const float* resid,
                               float* Cf, unsigned short* Cbf,
                               int M, int N, int K, int lda, int ldb, int ldc,
                               float alpha, int gelu) {
  if (N >= 256) {   // macro-tiled path (M is a multiple of 64 in all uses)
    dim3 grid((N + 255) / 256, M / 64);
    gemm_nt_v2_k<<<grid, 256, 0, s>>>(A, B, bias, resid, Cf, Cbf,
                                      M, N, K, lda, ldb, ldc, alpha, gelu);
  } else {          // small-N path (e.g. P@V with N=64)
    dim3 grid((N / 16 + 7) / 8, M / 16);
    gemm_nt_k<<<grid, 256, 0, s>>>(A, B, bias, resid, Cf, Cbf,
                                   M, N, K, lda, ldb, ldc, alpha, gelu);
  }
}

extern "C" void kernel_launch(void* const* d_in, const int* in_sizes, int n_in,
                              void* d_out, int out_size, void* d_ws, size_t ws_size,
                              hipStream_t stream) {
  (void)in_sizes; (void)n_in; (void)out_size; (void)ws_size;

  constexpr int V = 32000, S = 2048, Dm = 1024, H = 16, HD = 64, L = 4, FF = 4096;

  const int*   idx   = (const int*)  d_in[0];
  const float* wte   = (const float*)d_in[1];
  const float* wpe   = (const float*)d_in[2];
  const float* ln1w  = (const float*)d_in[3];
  const float* ln1b  = (const float*)d_in[4];
  const float* wqkv  = (const float*)d_in[5];
  const float* bqkv  = (const float*)d_in[6];
  const float* wo    = (const float*)d_in[7];
  const float* bo    = (const float*)d_in[8];
  const float* ln2w  = (const float*)d_in[9];
  const float* ln2b  = (const float*)d_in[10];
  const float* wfc   = (const float*)d_in[11];
  const float* bfc   = (const float*)d_in[12];
  const float* wfc2  = (const float*)d_in[13];
  const float* bfc2  = (const float*)d_in[14];
  const float* lnfw  = (const float*)d_in[15];
  const float* lnfb  = (const float*)d_in[16];
  float* logits = (float*)d_out;

  // ---- workspace carve (bump allocator, 256B aligned) ----
  char* p = (char*)d_ws;
  auto alloc = [&](size_t bytes) -> char* {
    char* r = p; p += (bytes + 255) & ~(size_t)255; return r;
  };
  unsigned short* wqkvT = (unsigned short*)alloc((size_t)L * 3 * Dm * Dm * 2);
  unsigned short* woT   = (unsigned short*)alloc((size_t)L * Dm * Dm * 2);
  unsigned short* wfcT  = (unsigned short*)alloc((size_t)L * FF * Dm * 2);
  unsigned short* wfc2T = (unsigned short*)alloc((size_t)L * Dm * FF * 2);
  unsigned short* wteb  = (unsigned short*)alloc((size_t)V * Dm * 2);
  float*          xf    = (float*)         alloc((size_t)S * Dm * 4);
  unsigned short* hbf   = (unsigned short*)alloc((size_t)S * Dm * 2);
  float*          qkvf  = (float*)         alloc((size_t)S * 3 * Dm * 4);
  unsigned short* qbf   = (unsigned short*)alloc((size_t)S * Dm * 2);
  unsigned short* kbf   = (unsigned short*)alloc((size_t)S * Dm * 2);
  unsigned short* vTbf  = (unsigned short*)alloc((size_t)S * Dm * 2);
  float*          scf   = (float*)         alloc((size_t)S * S * 4);
  unsigned short* prbf  = (unsigned short*)alloc((size_t)S * S * 2);
  unsigned short* obf   = (unsigned short*)alloc((size_t)S * Dm * 2);
  unsigned short* ffbf  = (unsigned short*)alloc((size_t)S * FF * 2);
  unsigned short* xbf   = (unsigned short*)alloc((size_t)S * Dm * 2);

  auto blocks = [](size_t n) { return (unsigned)((n + 255) / 256); };

  // ---- weight conversion to transposed bf16 ----
  conv_k<<<blocks((size_t)V * Dm), 256, 0, stream>>>(wte, wteb, (size_t)V * Dm);
  for (int l = 0; l < L; ++l) {
    tconv_k<<<blocks((size_t)Dm * 3 * Dm), 256, 0, stream>>>(
        wqkv + (size_t)l * Dm * 3 * Dm, wqkvT + (size_t)l * 3 * Dm * Dm, Dm, 3 * Dm);
    tconv_k<<<blocks((size_t)Dm * Dm), 256, 0, stream>>>(
        wo + (size_t)l * Dm * Dm, woT + (size_t)l * Dm * Dm, Dm, Dm);
    tconv_k<<<blocks((size_t)Dm * FF), 256, 0, stream>>>(
        wfc + (size_t)l * Dm * FF, wfcT + (size_t)l * FF * Dm, Dm, FF);
    tconv_k<<<blocks((size_t)FF * Dm), 256, 0, stream>>>(
        wfc2 + (size_t)l * FF * Dm, wfc2T + (size_t)l * Dm * FF, FF, Dm);
  }

  // ---- embeddings ----
  embed_k<<<blocks((size_t)S * Dm), 256, 0, stream>>>(idx, wte, wpe, xf, S, Dm);

  // ---- transformer blocks ----
  for (int l = 0; l < L; ++l) {
    layernorm_k<<<S, 256, 0, stream>>>(xf, ln1w + l * Dm, ln1b + l * Dm, hbf, Dm);
    launch_gemm(stream, hbf, wqkvT + (size_t)l * 3 * Dm * Dm, bqkv + l * 3 * Dm,
                nullptr, qkvf, nullptr, S, 3 * Dm, Dm, Dm, Dm, 3 * Dm, 1.f, 0);
    split_qkv_k<<<blocks((size_t)S * Dm), 256, 0, stream>>>(qkvf, qbf, kbf, vTbf,
                                                            S, Dm, HD);

    for (int h = 0; h < H; ++h) {
      const unsigned short* qh = qbf  + (size_t)h * S * HD;
      const unsigned short* kh = kbf  + (size_t)h * S * HD;
      const unsigned short* vh = vTbf + (size_t)h * HD * S;
      // scores = (Q @ K^T) / sqrt(HD)
      launch_gemm(stream, qh, kh, nullptr, nullptr, scf, nullptr,
                  S, S, HD, HD, HD, S, 0.125f, 0);
      softmax_causal_k<<<S, 256, 0, stream>>>(scf, prbf, S);
      // o_head = P @ V  -> bf16 slice of o [S, D] at column h*HD
      launch_gemm(stream, prbf, vh, nullptr, nullptr, nullptr, obf + h * HD,
                  S, HD, S, S, S, Dm, 1.f, 0);
    }

    // x = x + o @ Wo + bo
    launch_gemm(stream, obf, woT + (size_t)l * Dm * Dm, bo + l * Dm,
                xf, xf, nullptr, S, Dm, Dm, Dm, Dm, Dm, 1.f, 0);
    layernorm_k<<<S, 256, 0, stream>>>(xf, ln2w + l * Dm, ln2b + l * Dm, hbf, Dm);
    // ff = gelu(h @ Wfc + bfc)
    launch_gemm(stream, hbf, wfcT + (size_t)l * FF * Dm, bfc + l * FF,
                nullptr, nullptr, ffbf, S, FF, Dm, Dm, Dm, FF, 1.f, 1);
    // x = x + ff @ Wfc2 + bfc2
    launch_gemm(stream, ffbf, wfc2T + (size_t)l * Dm * FF, bfc2 + l * Dm,
                xf, xf, nullptr, S, Dm, FF, FF, FF, Dm, 1.f, 0);
  }

  // ---- final LN + weight-tied lm_head ----
  layernorm_k<<<S, 256, 0, stream>>>(xf, lnfw, lnfb, xbf, Dm);
  launch_gemm(stream, xbf, wteb, nullptr, nullptr, logits, nullptr,
              S, V, Dm, Dm, Dm, V, 1.f, 0);
}
